// SparseSelfAttention_11175504904723
// MI455X (gfx1250) — matrix operand order repaired and checked by reading, not compiled, not run on hardware
//
#include <hip/hip_runtime.h>
#include <hip/hip_bf16.h>
#include <math.h>

// Problem constants (from reference)
#define Bm   2
#define Lm   2048
#define Dm   256
#define Hh   8
#define HDm  32
#define KNN  30
#define NROWS (Bm*Lm)          // 4096
#define EPSc 1e-6f
#define LN_EPSc 1e-5f

typedef __attribute__((ext_vector_type(16))) __bf16 v16bf;
typedef __attribute__((ext_vector_type(8)))  float  v8f;
typedef __attribute__((ext_vector_type(4)))  unsigned int u32x4;

// ---------------------------------------------------------------------------
// Kernel 0: fp32 -> bf16 conversion (one-shot; halves GEMM read traffic)
// ---------------------------------------------------------------------------
__global__ void cvt_bf16_kernel(const float* __restrict__ src,
                                __bf16* __restrict__ dst, int n) {
    int i = blockIdx.x * blockDim.x + threadIdx.x;
    if (i < n) dst[i] = (__bf16)src[i];
}

// ---------------------------------------------------------------------------
// Kernel 1: QKV projections via v_wmma_f32_16x16x32_bf16.
// out[n,m] = sum_k x[n,k] * W[m,k] + bias[m]   (x @ W.T + b)
// One wave computes a 16x64 output tile: 4 accumulators share each A fetch
// (2.5 b128 loads per WMMA, 32 WMMAs per wave). K=256 -> 8 K-steps.
// A layout (16-bit A 16x32): lane<16 row M=lane, VGPR v holds K pair
//   c_v = 2v + (v>=4 ? 8 : 0); lanes>=16 same rows, +8 K offset.
//   => per lane: two contiguous 16B loads at col (kk + hi*8) and (kk+16+hi*8).
// B layout (16-bit B 32x16): lane<16 col N=lane, K=0..7,16..23 across 8 VGPRs;
//   lanes>=16 K=8..15,24..31. => one row of W, two 16B loads at kk + hi*8 and kk+16+hi*8.
// ---------------------------------------------------------------------------
__global__ void qkv_gemm_kernel(const __bf16* __restrict__ xb,     // [4096,256]
                                const __bf16* __restrict__ Wb,     // [3,256,256]
                                const float* __restrict__ bq,
                                const float* __restrict__ bk,
                                const float* __restrict__ bv,
                                float* __restrict__ qkv) {         // [3,4096,256]
    const int gwave = (blockIdx.x * blockDim.x + threadIdx.x) >> 5;
    const int lane  = threadIdx.x & 31;
    const int tilesPerMat = (NROWS / 16) * (Dm / 64);   // 256*4 = 1024
    const int mat = gwave / tilesPerMat;                // 0=Q,1=K,2=V
    const int rem = gwave % tilesPerMat;
    const int mt  = rem / (Dm / 64);                    // M tile (rows of x)
    const int ng  = rem % (Dm / 64);                    // N group (4 x 16 cols)

    const __bf16* W    = Wb + (size_t)mat * Dm * Dm;
    const float*  bias = (mat == 0) ? bq : (mat == 1) ? bk : bv;
    float*        Out  = qkv + (size_t)mat * NROWS * Dm;

    const int hi   = lane >> 4;          // half-wave select
    const int l16  = lane & 15;
    const int arow = mt * 16 + l16;      // x row this lane supplies

    v8f acc[4] = {v8f{}, v8f{}, v8f{}, v8f{}};
    union { u32x4 u[2]; v16bf v; } A, Bt;

#pragma unroll
    for (int kk = 0; kk < Dm; kk += 32) {
        const __bf16* ap = xb + (size_t)arow * Dm + kk + (hi ? 8 : 0);
        A.u[0] = *(const u32x4*)(ap);          // K pairs 2v   (+hi*8)
        A.u[1] = *(const u32x4*)(ap + 16);     // K pairs 2v+8 (+hi*8)
#pragma unroll
        for (int s = 0; s < 4; s++) {
            const int brow = ng * 64 + s * 16 + l16;   // W row == output col
            const __bf16* bp = W + (size_t)brow * Dm + kk + (hi ? 8 : 0);
            Bt.u[0] = *(const u32x4*)(bp);      // K 0..7   (+hi*8)
            Bt.u[1] = *(const u32x4*)(bp + 16); // K 16..23 (+hi*8)
            acc[s] = __builtin_amdgcn_wmma_f32_16x16x32_bf16(
                         false, A.v, false, Bt.v, (short)0, acc[s], false, false);
        }
    }

    // C/D layout: VGPR i -> row (mt*16 + i + hi*8), col (base + l16)
#pragma unroll
    for (int s = 0; s < 4; s++) {
        const int col  = ng * 64 + s * 16 + l16;
        const float bc = bias[col];
#pragma unroll
        for (int i = 0; i < 8; i++) {
            const int row = mt * 16 + i + (hi ? 8 : 0);
            Out[(size_t)row * Dm + col] = acc[s][i] + bc;
        }
    }
}

// ---------------------------------------------------------------------------
// Kernel 2: top-30 neighbors per (b,q) row of distance_weights [B,L,L].
// One 256-thread block per row. Row is staged memory->LDS with CDNA5
// GLOBAL_LOAD_ASYNC_TO_LDS_B128 (ASYNCcnt path, no VGPR round-trip), then
// 30x iterative argmax with lowest-index tie-break (matches jax.lax.top_k).
// ---------------------------------------------------------------------------
__global__ void topk_kernel(const float* __restrict__ dw,
                            int*   __restrict__ topi,   // [4096,32]
                            float* __restrict__ topv) { // [4096,32]
    __shared__ float vals[Lm];
    __shared__ float rv[256];
    __shared__ int   ri[256];
    const int row = blockIdx.x;
    const int tid = threadIdx.x;
    const float* p = dw + (size_t)row * Lm;

    // Async stage 8KB row into LDS: 2 passes x 256 lanes x 16B.
#pragma unroll
    for (int pass = 0; pass < Lm / (256 * 4); pass++) {
        const float* gsrc = p + pass * 1024 + tid * 4;
        unsigned ldsb = (unsigned)(size_t)(&vals[pass * 1024 + tid * 4]);
        asm volatile("global_load_async_to_lds_b128 %0, %1, off"
                     :: "v"(ldsb), "v"(gsrc) : "memory");
    }
    asm volatile("s_wait_asynccnt 0x0" ::: "memory");
    __syncthreads();

    for (int sel = 0; sel < KNN; sel++) {
        float bestv = -INFINITY; int besti = 0x7fffffff;
#pragma unroll
        for (int t = 0; t < Lm / 256; t++) {
            const int idx = t * 256 + tid;          // ascending: keeps lowest idx
            const float v = vals[idx];
            if (v > bestv) { bestv = v; besti = idx; }
        }
        rv[tid] = bestv; ri[tid] = besti;
        __syncthreads();
        for (int s = 128; s > 0; s >>= 1) {
            if (tid < s) {
                const float v2 = rv[tid + s]; const int i2 = ri[tid + s];
                if (v2 > rv[tid] || (v2 == rv[tid] && i2 < ri[tid])) {
                    rv[tid] = v2; ri[tid] = i2;
                }
            }
            __syncthreads();
        }
        if (tid == 0) {
            const int bi = ri[0];
            topi[row * 32 + sel] = bi;
            topv[row * 32 + sel] = rv[0];
            vals[bi] = -INFINITY;                   // remove from next rounds
        }
        __syncthreads();
    }
}

// ---------------------------------------------------------------------------
// Kernel 3: gathered 30-neighbor attention + residual + LayerNorm, fused.
// One block per (b,q); wave h handles head h (wave32: lane d = HD element d).
// ---------------------------------------------------------------------------
__global__ void attn_ln_kernel(const float* __restrict__ Q,
                               const float* __restrict__ Kb,
                               const float* __restrict__ Vb,
                               const float* __restrict__ x,
                               const int*   __restrict__ topi,
                               const float* __restrict__ topv,
                               const float* __restrict__ dbw,
                               const float* __restrict__ ln_g,
                               const float* __restrict__ ln_b,
                               float* __restrict__ out) {
    __shared__ float s1[Hh], s2[Hh];
    const int row  = blockIdx.x;          // b*L + q
    const int b    = row >> 11;           // /2048
    const int tid  = threadIdx.x;         // == feature index h*32+lane
    const int lane = tid & 31;
    const int h    = tid >> 5;

    int   myidx = 0;
    float mydw  = 0.f;
    if (lane < KNN) {
        myidx = topi[row * 32 + lane];
        mydw  = topv[row * 32 + lane];
    }
    const float qd    = Q[(size_t)row * Dm + tid];
    const float scale = 0.1767766952966369f;   // 1/sqrt(32)
    const float bw    = dbw[0];

    // scores: lane j ends up holding score_j (j < 30)
    float my_score = -INFINITY;
    for (int j = 0; j < KNN; j++) {
        const int kidx = __shfl(myidx, j, 32);
        const float kv = Kb[((size_t)(b * Lm + kidx)) * Dm + tid]; // coalesced
        float part = qd * kv;
#pragma unroll
        for (int off = 16; off > 0; off >>= 1) part += __shfl_xor(part, off, 32);
        if (lane == j)
            my_score = part * scale + bw * __logf(fmaxf(mydw, EPSc));
    }

    // softmax across lanes 0..29
    float m = my_score;
#pragma unroll
    for (int off = 16; off > 0; off >>= 1) m = fmaxf(m, __shfl_xor(m, off, 32));
    const float e = (lane < KNN) ? __expf(my_score - m) : 0.f;
    float ssum = e;
#pragma unroll
    for (int off = 16; off > 0; off >>= 1) ssum += __shfl_xor(ssum, off, 32);
    const float prob = e / ssum;

    // out_d = sum_j p_j * V[idx_j][d]   (coalesced V reads)
    float acc = 0.f;
    for (int j = 0; j < KNN; j++) {
        const float pj  = __shfl(prob, j, 32);
        const int  kidx = __shfl(myidx, j, 32);
        acc = fmaf(pj, Vb[((size_t)(b * Lm + kidx)) * Dm + tid], acc);
    }

    // residual + LayerNorm over 256 features (8 wave partials via LDS)
    const float hval = acc + x[(size_t)row * Dm + tid];
    float a1 = hval, a2 = hval * hval;
#pragma unroll
    for (int off = 16; off > 0; off >>= 1) {
        a1 += __shfl_xor(a1, off, 32);
        a2 += __shfl_xor(a2, off, 32);
    }
    if (lane == 0) { s1[h] = a1; s2[h] = a2; }
    __syncthreads();
    float t1 = 0.f, t2 = 0.f;
#pragma unroll
    for (int w = 0; w < Hh; w++) { t1 += s1[w]; t2 += s2[w]; }
    const float mean = t1 * (1.f / Dm);
    const float var  = t2 * (1.f / Dm) - mean * mean;
    const float inv  = rsqrtf(var + LN_EPSc);
    out[(size_t)row * Dm + tid] = (hval - mean) * inv * ln_g[tid] + ln_b[tid];
}

// ---------------------------------------------------------------------------
extern "C" void kernel_launch(void* const* d_in, const int* in_sizes, int n_in,
                              void* d_out, int out_size, void* d_ws, size_t ws_size,
                              hipStream_t stream) {
    const float* x   = (const float*)d_in[0];
    const float* dw  = (const float*)d_in[1];
    const float* Wq  = (const float*)d_in[2];
    const float* bq  = (const float*)d_in[3];
    const float* Wk  = (const float*)d_in[4];
    const float* bk  = (const float*)d_in[5];
    const float* Wv  = (const float*)d_in[6];
    const float* bv  = (const float*)d_in[7];
    const float* dbw = (const float*)d_in[8];
    const float* lng = (const float*)d_in[9];
    const float* lnb = (const float*)d_in[10];
    float* out = (float*)d_out;

    // Workspace layout
    char* ws = (char*)d_ws;
    size_t off = 0;
    __bf16* xb = (__bf16*)(ws + off); off += (size_t)NROWS * Dm * 2;        // 2 MB
    __bf16* Wb = (__bf16*)(ws + off); off += (size_t)3 * Dm * Dm * 2;       // 384 KB
    off = (off + 255) & ~(size_t)255;
    float* qkv = (float*)(ws + off);  off += (size_t)3 * NROWS * Dm * 4;    // 12 MB
    float* Qp = qkv;
    float* Kp = qkv + (size_t)NROWS * Dm;
    float* Vp = qkv + (size_t)2 * NROWS * Dm;
    int*   topi = (int*)(ws + off);   off += (size_t)NROWS * 32 * 4;        // 512 KB
    float* topv = (float*)(ws + off); off += (size_t)NROWS * 32 * 4;        // 512 KB

    // 0) fp32 -> bf16 staging
    cvt_bf16_kernel<<<(NROWS * Dm + 255) / 256, 256, 0, stream>>>(x, xb, NROWS * Dm);
    cvt_bf16_kernel<<<(Dm * Dm + 255) / 256, 256, 0, stream>>>(Wq, Wb,               Dm * Dm);
    cvt_bf16_kernel<<<(Dm * Dm + 255) / 256, 256, 0, stream>>>(Wk, Wb + Dm * Dm,     Dm * Dm);
    cvt_bf16_kernel<<<(Dm * Dm + 255) / 256, 256, 0, stream>>>(Wv, Wb + 2 * Dm * Dm, Dm * Dm);

    // 1) Q/K/V projections: 3 * 256 * 4 = 3072 wave-tiles (16x64 each)
    qkv_gemm_kernel<<<3072 / 8, 256, 0, stream>>>(xb, Wb, bq, bk, bv, qkv);

    // 2) top-30 neighbor selection per (b,q)
    topk_kernel<<<NROWS, 256, 0, stream>>>(dw, topi, topv);

    // 3) sparse attention + residual + LayerNorm (fused)
    attn_ln_kernel<<<NROWS, 256, 0, stream>>>(Qp, Kp, Vp, x, topi, topv,
                                              dbw, lng, lnb, out);
}
